// CloudTransformerMM_15023795601612
// MI455X (gfx1250) — matrix operand order repaired and checked by reading, not compiled, not run on hardware
//
#include <hip/hip_runtime.h>
#include <hip/hip_bf16.h>

// ---------------------------------------------------------------------------
// CDNA5 (gfx1250) transformer forward pass: bf16 WMMA GEMMs + flash attention
// ---------------------------------------------------------------------------

typedef __attribute__((ext_vector_type(16))) __bf16 bf16x16;
typedef __attribute__((ext_vector_type(8)))  __bf16 bf16x8;
typedef __attribute__((ext_vector_type(8)))  float  f32x8;

// Model dimensions (fixed by the reference)
constexpr int B_   = 2;
constexpr int S_   = 2048;
constexpr int D_   = 1024;
constexpr int NH_  = 16;
constexpr int KVH_ = 4;
constexpr int HD_  = 64;
constexpr int NREP_ = NH_ / KVH_;
constexpr int L_   = 8;
constexpr int DFF_ = 4096;
constexpr int V_   = 32000;
constexpr int M_   = B_ * S_;     // 4096 token rows
constexpr float EPS_    = 1e-6f;
constexpr float THETA_  = 10000.0f;
constexpr float YSCALE_ = 40.0f;
constexpr float YALPHA_ = 1.0f;
constexpr float YBETA_  = 32.0f;
constexpr float ROPE_MAX_ = 2048.0f * 40.0f;

static __device__ __forceinline__ f32x8 wmma_bf16(bf16x16 a, bf16x16 b, f32x8 c) {
  return __builtin_amdgcn_wmma_f32_16x16x32_bf16(false, a, false, b, (short)0, c,
                                                 false, false);
}

static __device__ __forceinline__ bf16x16 pack16(bf16x8 lo, bf16x8 hi) {
  bf16x16 r;
#pragma unroll
  for (int i = 0; i < 8; ++i) { r[i] = lo[i]; r[8 + i] = hi[i]; }
  return r;
}

// A-fragment loader: p points at (row, kblock + akoff) in a bf16 row-major matrix.
// Lane half 0 takes K {0..15}, half 1 takes K {16..31}.
static __device__ __forceinline__ bf16x16 load_a(const __bf16* p) {
  return pack16(*reinterpret_cast<const bf16x8*>(p),
                *reinterpret_cast<const bf16x8*>(p + 8));
}

// ---------------------------------------------------------------------------
// Elementwise / prep kernels
// ---------------------------------------------------------------------------

__global__ void k_f32_to_bf16(const float* __restrict__ s, __bf16* __restrict__ d,
                              size_t n) {
  size_t i = (size_t)blockIdx.x * blockDim.x + threadIdx.x;
  if (i < n) d[i] = (__bf16)s[i];
}

__global__ void k_embed(const int* __restrict__ ids, const float* __restrict__ emb,
                        float* __restrict__ h) {
  size_t i = (size_t)blockIdx.x * blockDim.x + threadIdx.x;
  if (i < (size_t)M_ * D_) {
    size_t r = i / D_;
    size_t c = i - r * D_;
    h[i] = emb[(size_t)ids[r] * D_ + c];
  }
}

__global__ __launch_bounds__(256) void k_rmsnorm(const float* __restrict__ x,
                                                 const float* __restrict__ w,
                                                 __bf16* __restrict__ out) {
  __shared__ float red[8];
  const size_t row = blockIdx.x;
  const float* xr = x + row * D_;
  float ss = 0.0f;
  for (int i = threadIdx.x; i < D_; i += 256) { float v = xr[i]; ss += v * v; }
#pragma unroll
  for (int m = 16; m >= 1; m >>= 1) ss += __shfl_xor(ss, m);
  if ((threadIdx.x & 31) == 0) red[threadIdx.x >> 5] = ss;
  __syncthreads();
  float tot = 0.0f;
#pragma unroll
  for (int i = 0; i < 8; ++i) tot += red[i];
  const float rinv = rsqrtf(tot * (1.0f / D_) + EPS_);
  __bf16* orow = out + row * D_;
  for (int i = threadIdx.x; i < D_; i += 256)
    orow[i] = (__bf16)(xr[i] * rinv * w[i]);
}

// RoPE with the reference's YaRN-style frequency warp, computed inline.
// src: [M_, H*HD] (GEMM output, token-major); dst: [B*H, S, HD] (head-major).
__global__ void k_rope(const __bf16* __restrict__ src, __bf16* __restrict__ dst,
                       int H) {
  const size_t total = (size_t)M_ * H * HD_;
  size_t i = (size_t)blockIdx.x * blockDim.x + threadIdx.x;
  if (i >= total) return;
  const int d = (int)(i & (HD_ - 1));
  size_t t = i >> 6;                 // HD_ == 64
  const int h = (int)(t % H);
  const size_t r = t / H;            // token row = b*S + s
  const int s = (int)(r % S_);
  const int b = (int)(r / S_);

  const int fi = d & (HD_ / 2 - 1);
  float inv = __expf(-(2.0f * fi / HD_) * __logf(THETA_));
  const float wavelen = 6.28318530717958647692f / inv;
  const float rr = ROPE_MAX_ / wavelen;
  float gam = (rr - YALPHA_) / (YBETA_ - YALPHA_);
  gam = fminf(fmaxf(gam, 0.0f), 1.0f);
  inv = inv * ((1.0f - gam) / YSCALE_ + gam);
  const float ang = (s * inv) * rsqrtf(0.1f * __logf(YSCALE_) + 1.0f);
  float sn, cs;
  __sincosf(ang, &sn, &cs);

  const __bf16* base = src + r * (size_t)(H * HD_) + (size_t)h * HD_;
  const float x  = (float)base[d];
  const float xo = (d < HD_ / 2) ? -(float)base[d + HD_ / 2]
                                 :  (float)base[d - HD_ / 2];
  dst[(((size_t)(b * H + h)) * S_ + s) * HD_ + d] = (__bf16)(x * cs + xo * sn);
}

// V transpose: [M_, KVH*HD] -> [B*KVH, HD, S] so PV B-fragments are contiguous.
__global__ void k_vtrans(const __bf16* __restrict__ src, __bf16* __restrict__ dst) {
  const size_t total = (size_t)M_ * KVH_ * HD_;
  size_t i = (size_t)blockIdx.x * blockDim.x + threadIdx.x;
  if (i >= total) return;
  const int d = (int)(i & (HD_ - 1));
  size_t t = i >> 6;
  const int h = (int)(t % KVH_);
  const size_t r = t / KVH_;
  const int s = (int)(r % S_);
  const int b = (int)(r / S_);
  dst[(((size_t)(b * KVH_ + h)) * HD_ + d) * S_ + s] =
      src[r * (size_t)(KVH_ * HD_) + (size_t)h * HD_ + d];
}

// ---------------------------------------------------------------------------
// WMMA GEMM: C[M,N] = A[M,K] * B[N,K]^T   (A,B bf16 row-major, K-major dot)
// Register-blocked: each wave computes 2 m-tiles x 4 n-tiles (32x64 of C),
// reusing each A fragment 4x and each B fragment 2x (8 WMMAs / 6 loads per
// 32-deep K step). A 128-thread block covers a 32x256 tile of C.
// MODE 0: store bf16; MODE 1: f32 residual +=; MODE 2: store f32.
// ---------------------------------------------------------------------------
template <int MODE>
__global__ __launch_bounds__(128) void k_gemm(const __bf16* __restrict__ A,
                                              const __bf16* __restrict__ Bw,
                                              void* __restrict__ Cp,
                                              int N, int K) {
  const int lane = threadIdx.x & 31;
  const int wave = threadIdx.x >> 5;
  const int nt0 = (blockIdx.x * 4 + wave) * 4;   // first of 4 n-tiles
  const int mt0 = blockIdx.y * 2;                // first of 2 m-tiles
  const int hl = lane >> 4;                      // lane half
  const int l15 = lane & 15;
  const __bf16* ap0 = A + (size_t)(mt0 * 16 + l15) * K + (hl ? 16 : 0);
  const __bf16* ap1 = ap0 + (size_t)16 * K;
  const __bf16* bp = Bw + (size_t)(nt0 * 16 + l15) * K + (hl ? 16 : 0);
  const size_t bstride = (size_t)16 * K;
  f32x8 acc[2][4] = {};
  for (int kb = 0; kb < K; kb += 32) {
    const bf16x16 a0 = load_a(ap0 + kb);
    const bf16x16 a1 = load_a(ap1 + kb);
#pragma unroll
    for (int j = 0; j < 4; ++j) {
      __builtin_prefetch(bp + j * bstride + kb + 512, 0, 3);
      const bf16x16 bv =
          *reinterpret_cast<const bf16x16*>(bp + j * bstride + kb);
      acc[0][j] = wmma_bf16(a0, bv, acc[0][j]);
      acc[1][j] = wmma_bf16(a1, bv, acc[1][j]);
    }
  }
#pragma unroll
  for (int mi = 0; mi < 2; ++mi) {
#pragma unroll
    for (int j = 0; j < 4; ++j) {
      const int col = (nt0 + j) * 16 + l15;
#pragma unroll
      for (int i = 0; i < 8; ++i) {
        const int row = (mt0 + mi) * 16 + hl * 8 + i;
        const size_t idx = (size_t)row * N + col;
        if (MODE == 0)      ((__bf16*)Cp)[idx] = (__bf16)acc[mi][j][i];
        else if (MODE == 1) ((float*)Cp)[idx] += acc[mi][j][i];
        else                ((float*)Cp)[idx]  = acc[mi][j][i];
      }
    }
  }
}

// Fused gated-FFN GEMM: out = silu(A*B1^T) * (A*B3^T), bf16 output.
// Register-blocked 2 m-tiles x 2 n-tiles per wave over both weight matrices
// (8 WMMAs / 6 fragment loads per K step). Block covers 32x128 of the output.
__global__ __launch_bounds__(128) void k_gemm_dual_silu(
    const __bf16* __restrict__ A, const __bf16* __restrict__ B1,
    const __bf16* __restrict__ B3, __bf16* __restrict__ C, int N, int K) {
  const int lane = threadIdx.x & 31;
  const int wave = threadIdx.x >> 5;
  const int nt0 = (blockIdx.x * 4 + wave) * 2;   // first of 2 n-tiles
  const int mt0 = blockIdx.y * 2;                // first of 2 m-tiles
  const int hl = lane >> 4;
  const int l15 = lane & 15;
  const __bf16* ap0 = A + (size_t)(mt0 * 16 + l15) * K + (hl ? 16 : 0);
  const __bf16* ap1 = ap0 + (size_t)16 * K;
  const __bf16* b1p = B1 + (size_t)(nt0 * 16 + l15) * K + (hl ? 16 : 0);
  const __bf16* b3p = B3 + (size_t)(nt0 * 16 + l15) * K + (hl ? 16 : 0);
  const size_t bstride = (size_t)16 * K;
  f32x8 g1[2][2] = {};
  f32x8 g3[2][2] = {};
  for (int kb = 0; kb < K; kb += 32) {
    const bf16x16 a0 = load_a(ap0 + kb);
    const bf16x16 a1 = load_a(ap1 + kb);
#pragma unroll
    for (int j = 0; j < 2; ++j) {
      __builtin_prefetch(b1p + j * bstride + kb + 512, 0, 3);
      __builtin_prefetch(b3p + j * bstride + kb + 512, 0, 3);
      const bf16x16 b1v =
          *reinterpret_cast<const bf16x16*>(b1p + j * bstride + kb);
      const bf16x16 b3v =
          *reinterpret_cast<const bf16x16*>(b3p + j * bstride + kb);
      g1[0][j] = wmma_bf16(a0, b1v, g1[0][j]);
      g1[1][j] = wmma_bf16(a1, b1v, g1[1][j]);
      g3[0][j] = wmma_bf16(a0, b3v, g3[0][j]);
      g3[1][j] = wmma_bf16(a1, b3v, g3[1][j]);
    }
  }
#pragma unroll
  for (int mi = 0; mi < 2; ++mi) {
#pragma unroll
    for (int j = 0; j < 2; ++j) {
      const int col = (nt0 + j) * 16 + l15;
#pragma unroll
      for (int i = 0; i < 8; ++i) {
        const int row = (mt0 + mi) * 16 + hl * 8 + i;
        const float x = g1[mi][j][i];
        const float gate = x / (1.0f + __expf(-x));
        C[(size_t)row * N + col] = (__bf16)(gate * g3[mi][j][i]);
      }
    }
  }
}

// ---------------------------------------------------------------------------
// Flash attention, causal, GQA. One wave per 16-query tile per (b, head).
// q: [B*NH, S, HD], k: [B*KVH, S, HD], vt: [B*KVH, HD, S], out: [B*S, NH*HD].
// Scores via 2x2 chained WMMAs per 32-key block; P re-laid-out through LDS.
// ---------------------------------------------------------------------------
__global__ __launch_bounds__(32) void k_attn(const __bf16* __restrict__ q,
                                             const __bf16* __restrict__ k,
                                             const __bf16* __restrict__ vt,
                                             __bf16* __restrict__ out) {
  __shared__ __bf16 lds_p[16 * 32];
  const int lane = threadIdx.x;
  const int l15 = lane & 15;
  const int hl = lane >> 4;
  const int qt = blockIdx.x;            // query tile (S/16)
  const int bh = blockIdx.y;            // b * NH + head
  const int b = bh / NH_, hh = bh % NH_;
  const int g = hh / NREP_;             // kv head
  const __bf16* qbase = q  + (size_t)bh * S_ * HD_;
  const __bf16* kbase = k  + (size_t)(b * KVH_ + g) * S_ * HD_;
  const __bf16* vbase = vt + (size_t)(b * KVH_ + g) * HD_ * S_;

  // Q A-fragments for the two 32-wide d-blocks (HD = 64).
  const int arow = qt * 16 + l15;
  const int akoff = hl ? 16 : 0;
  const bf16x16 qa0 = load_a(qbase + (size_t)arow * HD_ + 0  + akoff);
  const bf16x16 qa1 = load_a(qbase + (size_t)arow * HD_ + 32 + akoff);

  f32x8 acc0 = {}, acc1 = {}, acc2 = {}, acc3 = {};
  float mrow[8], lrow[8];
#pragma unroll
  for (int i = 0; i < 8; ++i) { mrow[i] = -1e30f; lrow[i] = 0.0f; }

  const int bkoff = hl ? 16 : 0;
  const int kend = qt * 16 + 16;        // causal limit (exclusive)
  const float scale = 0.125f;           // HD^-0.5

  for (int kb = 0; kb < kend; kb += 32) {
    // ---- scores: two 16x16 tiles covering keys kb..kb+31 ----
    f32x8 s0 = {}, s1 = {};
    {
      const __bf16* kp0 = kbase + (size_t)(kb + l15) * HD_ + bkoff;
      const __bf16* kp1 = kbase + (size_t)(kb + 16 + l15) * HD_ + bkoff;
      s0 = wmma_bf16(qa0, *reinterpret_cast<const bf16x16*>(kp0), s0);
      s0 = wmma_bf16(qa1, *reinterpret_cast<const bf16x16*>(kp0 + 32), s0);
      s1 = wmma_bf16(qa0, *reinterpret_cast<const bf16x16*>(kp1), s1);
      s1 = wmma_bf16(qa1, *reinterpret_cast<const bf16x16*>(kp1 + 32), s1);
    }

    // ---- online softmax (rows striped: lane half owns 8 rows) ----
    const int key0 = kb + l15, key1 = kb + 16 + l15;
#pragma unroll
    for (int i = 0; i < 8; ++i) {
      const int row = qt * 16 + hl * 8 + i;
      float r0 = (key0 <= row) ? s0[i] * scale : -1e30f;
      float r1 = (key1 <= row) ? s1[i] * scale : -1e30f;
      float mx = fmaxf(r0, r1);
#pragma unroll
      for (int md = 8; md >= 1; md >>= 1) mx = fmaxf(mx, __shfl_xor(mx, md));
      const float mnew = fmaxf(mrow[i], mx);
      const float corr = __expf(mrow[i] - mnew);
      const float p0 = __expf(r0 - mnew);
      const float p1 = __expf(r1 - mnew);
      float sum = p0 + p1;
#pragma unroll
      for (int md = 8; md >= 1; md >>= 1) sum += __shfl_xor(sum, md);
      lrow[i] = lrow[i] * corr + sum;
      mrow[i] = mnew;
      acc0[i] *= corr; acc1[i] *= corr; acc2[i] *= corr; acc3[i] *= corr;
      const int mm = hl * 8 + i;
      lds_p[mm * 32 + l15]      = (__bf16)p0;
      lds_p[mm * 32 + 16 + l15] = (__bf16)p1;
    }
    __syncthreads();

    // ---- P (A-layout from LDS) x V (contiguous B-fragments from vt) ----
    const int pk = hl ? 16 : 0;
    const bf16x16 pa =
        pack16(*reinterpret_cast<const bf16x8*>(&lds_p[l15 * 32 + pk]),
               *reinterpret_cast<const bf16x8*>(&lds_p[l15 * 32 + 8 + pk]));
    const __bf16* vp = vbase + (size_t)l15 * S_ + kb + bkoff;
    acc0 = wmma_bf16(pa, *reinterpret_cast<const bf16x16*>(vp + (size_t)0 * 16 * S_), acc0);
    acc1 = wmma_bf16(pa, *reinterpret_cast<const bf16x16*>(vp + (size_t)1 * 16 * S_), acc1);
    acc2 = wmma_bf16(pa, *reinterpret_cast<const bf16x16*>(vp + (size_t)2 * 16 * S_), acc2);
    acc3 = wmma_bf16(pa, *reinterpret_cast<const bf16x16*>(vp + (size_t)3 * 16 * S_), acc3);
    __syncthreads();
  }

  // ---- epilogue: out[b, s, hh*HD + d] = acc / l ----
  __bf16* obase = out + ((size_t)b * S_ + (size_t)qt * 16) * (NH_ * HD_) +
                  (size_t)hh * HD_;
#pragma unroll
  for (int i = 0; i < 8; ++i) {
    const float il = 1.0f / lrow[i];
    const size_t r = (size_t)(hl * 8 + i) * (NH_ * HD_);
    obase[r + 0 * 16 + l15] = (__bf16)(acc0[i] * il);
    obase[r + 1 * 16 + l15] = (__bf16)(acc1[i] * il);
    obase[r + 2 * 16 + l15] = (__bf16)(acc2[i] * il);
    obase[r + 3 * 16 + l15] = (__bf16)(acc3[i] * il);
  }
}

// ---------------------------------------------------------------------------
// Host orchestration
// ---------------------------------------------------------------------------

extern "C" void kernel_launch(void* const* d_in, const int* in_sizes, int n_in,
                              void* d_out, int out_size, void* d_ws, size_t ws_size,
                              hipStream_t stream) {
  (void)in_sizes; (void)n_in; (void)out_size; (void)ws_size;
  const int*   ids   = (const int*)d_in[0];
  const float* tok   = (const float*)d_in[1];
  const float* wq    = (const float*)d_in[2];
  const float* wk    = (const float*)d_in[3];
  const float* wv    = (const float*)d_in[4];
  const float* wo    = (const float*)d_in[5];
  const float* n1w   = (const float*)d_in[6];
  const float* n2w   = (const float*)d_in[7];
  const float* w1    = (const float*)d_in[8];
  const float* w2    = (const float*)d_in[9];
  const float* w3    = (const float*)d_in[10];
  const float* fnw   = (const float*)d_in[11];
  float* outp = (float*)d_out;

  char* ws = (char*)d_ws;
  size_t off = 0;
  auto alloc = [&](size_t bytes) -> char* {
    char* p = ws + off;
    off += (bytes + 255) & ~(size_t)255;
    return p;
  };

  const size_t n_wq = (size_t)L_ * NH_ * HD_ * D_;
  const size_t n_wk = (size_t)L_ * KVH_ * HD_ * D_;
  const size_t n_wv = n_wk;
  const size_t n_wo = (size_t)L_ * D_ * NH_ * HD_;
  const size_t n_w1 = (size_t)L_ * DFF_ * D_;
  const size_t n_w2 = (size_t)L_ * D_ * DFF_;
  const size_t n_w3 = n_w1;
  const size_t n_emb = (size_t)V_ * D_;

  __bf16* wq_bf  = (__bf16*)alloc(2 * n_wq);
  __bf16* wk_bf  = (__bf16*)alloc(2 * n_wk);
  __bf16* wv_bf  = (__bf16*)alloc(2 * n_wv);
  __bf16* wo_bf  = (__bf16*)alloc(2 * n_wo);
  __bf16* w1_bf  = (__bf16*)alloc(2 * n_w1);
  __bf16* w2_bf  = (__bf16*)alloc(2 * n_w2);
  __bf16* w3_bf  = (__bf16*)alloc(2 * n_w3);
  __bf16* emb_bf = (__bf16*)alloc(2 * n_emb);

  float*  h     = (float*)alloc(4 * (size_t)M_ * D_);
  __bf16* xn    = (__bf16*)alloc(2 * (size_t)M_ * D_);
  __bf16* qlin  = (__bf16*)alloc(2 * (size_t)M_ * NH_ * HD_);
  __bf16* klin  = (__bf16*)alloc(2 * (size_t)M_ * KVH_ * HD_);
  __bf16* vlin  = (__bf16*)alloc(2 * (size_t)M_ * KVH_ * HD_);
  __bf16* qr    = (__bf16*)alloc(2 * (size_t)M_ * NH_ * HD_);
  __bf16* kr    = (__bf16*)alloc(2 * (size_t)M_ * KVH_ * HD_);
  __bf16* vtb   = (__bf16*)alloc(2 * (size_t)M_ * KVH_ * HD_);
  __bf16* attn  = (__bf16*)alloc(2 * (size_t)M_ * NH_ * HD_);
  __bf16* yb    = (__bf16*)alloc(2 * (size_t)M_ * D_);
  __bf16* gate  = (__bf16*)alloc(2 * (size_t)M_ * DFF_);
  __bf16* hn    = (__bf16*)alloc(2 * (size_t)M_ * D_);

  auto cvt = [&](const float* s, __bf16* d, size_t n) {
    k_f32_to_bf16<<<dim3((unsigned)((n + 255) / 256)), dim3(256), 0, stream>>>(s, d, n);
  };
  cvt(wq, wq_bf, n_wq);   cvt(wk, wk_bf, n_wk);
  cvt(wv, wv_bf, n_wv);   cvt(wo, wo_bf, n_wo);
  cvt(w1, w1_bf, n_w1);   cvt(w2, w2_bf, n_w2);
  cvt(w3, w3_bf, n_w3);   cvt(tok, emb_bf, n_emb);

  {
    const size_t n = (size_t)M_ * D_;
    k_embed<<<dim3((unsigned)((n + 255) / 256)), dim3(256), 0, stream>>>(ids, tok, h);
  }

  const dim3 blk128(128), blk256(256), blk32(32);
  for (int l = 0; l < L_; ++l) {
    const __bf16* lwq = wq_bf + (size_t)l * NH_ * HD_ * D_;
    const __bf16* lwk = wk_bf + (size_t)l * KVH_ * HD_ * D_;
    const __bf16* lwv = wv_bf + (size_t)l * KVH_ * HD_ * D_;
    const __bf16* lwo = wo_bf + (size_t)l * D_ * NH_ * HD_;
    const __bf16* lw1 = w1_bf + (size_t)l * DFF_ * D_;
    const __bf16* lw2 = w2_bf + (size_t)l * D_ * DFF_;
    const __bf16* lw3 = w3_bf + (size_t)l * DFF_ * D_;

    k_rmsnorm<<<dim3(M_), blk256, 0, stream>>>(h, n1w + (size_t)l * D_, xn);

    // C tile per block: 32 rows x 256 cols
    k_gemm<0><<<dim3(NH_ * HD_ / 256, M_ / 32), blk128, 0, stream>>>(xn, lwq, qlin, NH_ * HD_, D_);
    k_gemm<0><<<dim3(KVH_ * HD_ / 256, M_ / 32), blk128, 0, stream>>>(xn, lwk, klin, KVH_ * HD_, D_);
    k_gemm<0><<<dim3(KVH_ * HD_ / 256, M_ / 32), blk128, 0, stream>>>(xn, lwv, vlin, KVH_ * HD_, D_);

    {
      const size_t nq = (size_t)M_ * NH_ * HD_;
      const size_t nk = (size_t)M_ * KVH_ * HD_;
      k_rope<<<dim3((unsigned)((nq + 255) / 256)), blk256, 0, stream>>>(qlin, qr, NH_);
      k_rope<<<dim3((unsigned)((nk + 255) / 256)), blk256, 0, stream>>>(klin, kr, KVH_);
      k_vtrans<<<dim3((unsigned)((nk + 255) / 256)), blk256, 0, stream>>>(vlin, vtb);
    }

    k_attn<<<dim3(S_ / 16, B_ * NH_), blk32, 0, stream>>>(qr, kr, vtb, attn);

    k_gemm<1><<<dim3(D_ / 256, M_ / 32), blk128, 0, stream>>>(attn, lwo, h, D_, NH_ * HD_);

    k_rmsnorm<<<dim3(M_), blk256, 0, stream>>>(h, n2w + (size_t)l * D_, yb);

    // C tile per block: 32 rows x 128 cols
    k_gemm_dual_silu<<<dim3(DFF_ / 128, M_ / 32), blk128, 0, stream>>>(yb, lw1, lw3, gate, DFF_, D_);

    k_gemm<1><<<dim3(D_ / 256, M_ / 32), blk128, 0, stream>>>(gate, lw2, h, D_, DFF_);
  }

  k_rmsnorm<<<dim3(M_), blk256, 0, stream>>>(h, fnw, hn);
  // LM head: [4096 x 32000] = hn @ emb^T, fp32 straight into d_out.
  k_gemm<2><<<dim3(V_ / 256, M_ / 32), blk128, 0, stream>>>(hn, emb_bf, outp, V_, D_);
}
